// MultiHeadedAttention_17188459119165
// MI455X (gfx1250) — matrix operand order repaired and checked by reading, not compiled, not run on hardware
//
#include <hip/hip_runtime.h>
#include <hip/hip_bf16.h>

// ---------------------------------------------------------------------------
// Fused MultiHeadedAttention for MI455X (gfx1250, wave32, WMMA bf16 + TDM)
// B=4, S=2048, D=1024, H=16, dk=64
// ---------------------------------------------------------------------------

typedef __bf16 bf16;
typedef __attribute__((ext_vector_type(16))) __bf16 bf16x16;
typedef __attribute__((ext_vector_type(8)))  __bf16 bf16x8;
typedef __attribute__((ext_vector_type(8)))  float  f32x8;
typedef __attribute__((ext_vector_type(4)))  unsigned int uint32x4;
typedef __attribute__((ext_vector_type(8)))  int int32x8;
typedef __attribute__((ext_vector_type(4)))  int int32x4;

static constexpr int Bc = 4;
static constexpr int Sc = 2048;
static constexpr int Dc = 1024;
static constexpr int Hc = 16;
static constexpr int DKc = 64;

// ---- WMMA fragment helpers (layouts per CDNA5 ISA 7.12.2, wave32) ----------

__device__ __forceinline__ bf16x16 frag16(const bf16* p0, const bf16* p1) {
  union { bf16x16 v; bf16x8 h[2]; } u;
  u.h[0] = *(const bf16x8*)p0;
  u.h[1] = *(const bf16x8*)p1;
  return u.v;
}

// A-matrix 16x32: lane (g=lane/16, m=lane%16): halves [8g..8g+7],[16+8g..+7]
__device__ __forceinline__ bf16x16 fragA(const bf16* rowp, int g) {
  return frag16(rowp + 8 * g, rowp + 16 + 8 * g);
}

// B-matrix 32x16: halves [8g..8g+7],[16+8g..+7] of column n; kcontig = 32-elem K strip for column n
__device__ __forceinline__ bf16x16 fragB(const bf16* kcontig, int g) {
  return frag16(kcontig + 8 * g, kcontig + 16 + 8 * g);
}

__device__ __forceinline__ f32x8 wmma_bf16(bf16x16 a, bf16x16 b, f32x8 c) {
  return __builtin_amdgcn_wmma_f32_16x16x32_bf16(
      false, a, false, b, (short)0, c, false, false);
}

// ---- Tensor Data Mover: 2D tile load, D# per CDNA5 ISA --
__device__ __forceinline__ void tdm_load_2d(unsigned lds_addr, const void* gptr,
                                            unsigned data_size_code,  // 1=2B,2=4B
                                            unsigned tile_d0, unsigned tile_d1,
                                            unsigned long long stride_d0) {
  unsigned long long ga = (unsigned long long)(size_t)gptr;
  unsigned td0 = (unsigned)(stride_d0 > tile_d0 ? stride_d0 : tile_d0);
  unsigned td1 = tile_d1;
  uint32x4 g0;
  g0[0] = 1u;                                          // count=1, user mode
  g0[1] = lds_addr;                                    // LDS byte address
  g0[2] = (unsigned)(ga & 0xffffffffull);              // global_addr[31:0]
  g0[3] = (unsigned)((ga >> 32) & 0x01ffffffull) | (2u << 30);  // type=2
  int32x8 g1;
  g1[0] = (int)(data_size_code << 16);                 // data_size
  g1[1] = (int)((td0 & 0xffffu) << 16);                // tensor_dim0 lo
  g1[2] = (int)(((td0 >> 16) & 0xffffu) | ((td1 & 0xffffu) << 16));
  g1[3] = (int)(((td1 >> 16) & 0xffffu) | ((tile_d0 & 0xffffu) << 16));
  g1[4] = (int)(tile_d1 & 0xffffu);                    // tile_dim1, tile_dim2=0
  g1[5] = (int)(stride_d0 & 0xffffffffull);            // dim0 stride lo
  g1[6] = (int)((stride_d0 >> 32) & 0xffffull);        // dim0 stride hi
  g1[7] = 0;
  int32x4 gz = {0, 0, 0, 0};
#if __clang_major__ >= 23
  int32x8 gz8 = {0, 0, 0, 0, 0, 0, 0, 0};
  __builtin_amdgcn_tensor_load_to_lds(g0, g1, gz, gz, gz8, 0);
#else
  __builtin_amdgcn_tensor_load_to_lds(g0, g1, gz, gz, 0);
#endif
}

__device__ __forceinline__ unsigned lds_addr_of(const void* p) {
  return (unsigned)(size_t)p;  // low 32 bits of generic addr = LDS offset
}

// ---------------------------------------------------------------------------
// Kernel 0a: f32 -> bf16 conversion (activations)
// ---------------------------------------------------------------------------
__global__ void cvt_f32_to_bf16(const float* __restrict__ src,
                                bf16* __restrict__ dst, long n) {
  long i = (long)blockIdx.x * blockDim.x + threadIdx.x;
  long stride = (long)gridDim.x * blockDim.x;
  for (; i < n; i += stride) dst[i] = (bf16)src[i];
}

// ---------------------------------------------------------------------------
// Kernel 0b: weight transpose f32[K][N] -> bf16 Wt[N][K] (LDS tiled)
// ---------------------------------------------------------------------------
__global__ __launch_bounds__(256)
void transpose_w_kernel(const float* __restrict__ src, bf16* __restrict__ dst) {
  __shared__ float tile[32][33];
  const int bx = blockIdx.x * 32, by = blockIdx.y * 32;
  const int tx = threadIdx.x & 31, ty = threadIdx.x >> 5;  // 32 x 8
  for (int j = 0; j < 32; j += 8)
    tile[ty + j][tx] = src[(long)(by + ty + j) * Dc + bx + tx];
  __syncthreads();
  for (int j = 0; j < 32; j += 8)
    dst[(long)(bx + ty + j) * Dc + by + tx] = (bf16)tile[tx][ty + j];
}

// ---------------------------------------------------------------------------
// Kernel 1: projection GEMM  C = scale * A[8192,1024] @ W, W given as Wt[N][K].
// Workgroup 8 waves = 2(M) x 4(N); wave tile 64x64 (4x4 WMMA frags -> 128 acc
// VGPRs + 64 frag VGPRs, stays under the 256-VGPR window, 4x reuse of both A
// and B fragments).  Weight slab [256 cols x 32 k] arrives by TDM, double-
// buffered; A-fragments direct from global (b128, K-contiguous).
//   mode 0: out[b][h][s][d]   (Q, K)    mode 1: out[b][h][d][s]  (V transposed)
// ---------------------------------------------------------------------------
__global__ __launch_bounds__(256)
void proj_gemm_kernel(const bf16* __restrict__ A, const bf16* __restrict__ Wt,
                      bf16* __restrict__ out, float scale, int mode) {
  __shared__ __align__(16) bf16 wt[2][256 * 32];  // [buf][lc][k], 2x16KB
  const int tid = threadIdx.x;
  const int wid = tid >> 5;
  const int lane = tid & 31;
  const int g = lane >> 4, n = lane & 15;
  const int wm = wid >> 2, wn = wid & 3;
  const int rowbase = blockIdx.x * 128 + wm * 64;
  const int colbase = blockIdx.y * 256;
  const int colw = wn * 64;

  f32x8 acc[4][4];
  for (int i = 0; i < 4; ++i)
    for (int j = 0; j < 4; ++j)
      for (int r = 0; r < 8; ++r) acc[i][j][r] = 0.f;

  constexpr int T = Dc / 32;  // 32 slabs
  const bf16* wsrc = Wt + (long)colbase * Dc;  // Wt rows = output columns
  if (wid == 0)
    tdm_load_2d(lds_addr_of(&wt[0][0]), wsrc, 1, 32, 256, Dc);

  for (int t = 0; t < T; ++t) {
    const int kk = t * 32;
    if (wid == 0) {
      if (t + 1 < T) {
        tdm_load_2d(lds_addr_of(&wt[(t + 1) & 1][0]), wsrc + kk + 32, 1, 32, 256, Dc);
        __builtin_amdgcn_s_wait_tensorcnt(1);  // slab t complete (in-order)
      } else {
        __builtin_amdgcn_s_wait_tensorcnt(0);
      }
    }
    __syncthreads();  // slab t visible to all waves

    const bf16* wbuf = &wt[t & 1][0];
    bf16x16 af[4], bfr[4];
    for (int fm = 0; fm < 4; ++fm)
      af[fm] = fragA(A + (long)(rowbase + fm * 16 + n) * Dc + kk, g);
    for (int fn = 0; fn < 4; ++fn)
      bfr[fn] = fragB(wbuf + (colw + fn * 16 + n) * 32, g);
    for (int fm = 0; fm < 4; ++fm)
      for (int fn = 0; fn < 4; ++fn)
        acc[fm][fn] = wmma_bf16(af[fm], bfr[fn], acc[fm][fn]);
    __syncthreads();  // all reads of buf t done before TDM reuses it
  }

  for (int fm = 0; fm < 4; ++fm)
    for (int fn = 0; fn < 4; ++fn)
      for (int r = 0; r < 8; ++r) {
        int row = rowbase + fm * 16 + 8 * g + r;   // b*S + s
        int col = colbase + colw + fn * 16 + n;    // h*dk + d
        int b = row >> 11, s = row & (Sc - 1);
        int h = col >> 6, d = col & (DKc - 1);
        long idx = (mode == 0)
            ? ((((long)b * Hc + h) * Sc + s) * DKc + d)
            : ((((long)b * Hc + h) * DKc + d) * Sc + s);
        out[idx] = (bf16)(acc[fm][fn][r] * scale);
      }
}

// ---------------------------------------------------------------------------
// Kernel 2: flash attention.  One workgroup = 4 waves = 64 query rows of one
// (b,h).  Per 32-key step, three TDM descriptors (K tile, Vt tile, mask tile)
// land in double-buffered LDS while the waves compute the previous step.
// Scores never touch global memory; Q is pre-scaled by 1/sqrt(dk).
// ---------------------------------------------------------------------------
__global__ __launch_bounds__(128)
void flash_attn_kernel(const bf16* __restrict__ Q, const bf16* __restrict__ K,
                       const bf16* __restrict__ Vt, const float* __restrict__ mask,
                       bf16* __restrict__ ctx) {
  __shared__ __align__(16) bf16 Ktile[2][32 * DKc];   // [buf][krow][d]   2x4KB
  __shared__ __align__(16) bf16 Vtile[2][DKc * 32];   // [buf][d][kcol]   2x4KB
  __shared__ __align__(16) float Mtile[2][64 * 32];   // [buf][qrow][kcol] 2x8KB
  __shared__ __align__(16) bf16 lds_p[4][16 * 32];    // per-wave P tile   4KB
  const int tid = threadIdx.x;
  const int wid = tid >> 5;
  const int lane = tid & 31;
  const int g = lane >> 4, n = lane & 15;
  const int bh = blockIdx.y;         // b*H + h
  const int b = bh >> 4, h = bh & 15;
  const int q0blk = blockIdx.x * 64;
  const int q0 = q0blk + wid * 16;

  const bf16* Qp = Q + ((long)bh * Sc + q0) * DKc;
  const bf16* Kp = K + (long)bh * Sc * DKc;
  const bf16* Vp = Vt + (long)bh * DKc * Sc;
  const float* mp = mask + (long)b * Sc * Sc + (long)q0blk * Sc;

  bf16x16 aq[2];
  {
    const bf16* p = Qp + (long)n * DKc;  // A row m = n
    aq[0] = fragA(p, g);                 // d = 0..31
    aq[1] = fragA(p + 32, g);            // d = 32..63
  }

  f32x8 accO[4];
  for (int df = 0; df < 4; ++df)
    for (int r = 0; r < 8; ++r) accO[df][r] = 0.f;
  float mrun[8], lrun[8];
  for (int r = 0; r < 8; ++r) { mrun[r] = -1e30f; lrun[r] = 0.f; }

  constexpr int T = Sc / 32;  // 64 steps

  if (wid == 0) {  // prologue: tile 0
    tdm_load_2d(lds_addr_of(&Ktile[0][0]), Kp, 1, DKc, 32, DKc);
    tdm_load_2d(lds_addr_of(&Vtile[0][0]), Vp, 1, 32, DKc, Sc);
    tdm_load_2d(lds_addr_of(&Mtile[0][0]), mp, 2, 32, 64, Sc);
  }

  for (int t = 0; t < T; ++t) {
    if (wid == 0) {
      if (t + 1 < T) {
        const int kn = (t + 1) * 32;
        const int nb = (t + 1) & 1;
        tdm_load_2d(lds_addr_of(&Ktile[nb][0]), Kp + (long)kn * DKc, 1, DKc, 32, DKc);
        tdm_load_2d(lds_addr_of(&Vtile[nb][0]), Vp + kn, 1, 32, DKc, Sc);
        tdm_load_2d(lds_addr_of(&Mtile[nb][0]), mp + kn, 2, 32, 64, Sc);
        __builtin_amdgcn_s_wait_tensorcnt(3);  // tile t's 3 loads complete
      } else {
        __builtin_amdgcn_s_wait_tensorcnt(0);
      }
    }
    __syncthreads();  // tile t visible

    const int buf = t & 1;
    // ---- scores: two 16-col tiles --------------------------------------
    f32x8 s[2];
    for (int nf = 0; nf < 2; ++nf) {
      const bf16* kp = &Ktile[buf][(nf * 16 + n) * DKc];
      bf16x16 bk0 = fragB(kp, g);        // d 0..31
      bf16x16 bk1 = fragB(kp + 32, g);   // d 32..63
      f32x8 c;
      for (int r = 0; r < 8; ++r) c[r] = 0.f;
      c = wmma_bf16(aq[0], bk0, c);
      c = wmma_bf16(aq[1], bk1, c);
      s[nf] = c;
    }
    // additive mask (from LDS tile)
    for (int nf = 0; nf < 2; ++nf)
      for (int r = 0; r < 8; ++r)
        s[nf][r] += Mtile[buf][(wid * 16 + 8 * g + r) * 32 + nf * 16 + n];

    // ---- online softmax ------------------------------------------------
    float mnew[8], alpha[8];
    for (int r = 0; r < 8; ++r) {
      float v = fmaxf(s[0][r], s[1][r]);
      for (int off = 1; off < 16; off <<= 1)
        v = fmaxf(v, __shfl_xor(v, off, 32));  // reduce over n-lanes
      mnew[r] = fmaxf(mrun[r], v);
      alpha[r] = __expf(mrun[r] - mnew[r]);
      mrun[r] = mnew[r];
    }
    float p0[8], p1[8];
    for (int r = 0; r < 8; ++r) {
      p0[r] = __expf(s[0][r] - mnew[r]);
      p1[r] = __expf(s[1][r] - mnew[r]);
      lrun[r] = lrun[r] * alpha[r] + p0[r] + p1[r];
    }
    for (int df = 0; df < 4; ++df)
      for (int r = 0; r < 8; ++r) accO[df][r] *= alpha[r];

    // ---- relayout P (C layout -> A layout) via per-wave LDS tile -------
    for (int r = 0; r < 8; ++r) {
      lds_p[wid][(8 * g + r) * 32 + n] = (bf16)p0[r];
      lds_p[wid][(8 * g + r) * 32 + 16 + n] = (bf16)p1[r];
    }
    asm volatile("s_wait_dscnt 0" ::: "memory");  // wave-local DS ordering
    bf16x16 ap = fragA(&lds_p[wid][n * 32], g);

    // ---- accO += P @ V -------------------------------------------------
    for (int df = 0; df < 4; ++df) {
      bf16x16 bv = fragB(&Vtile[buf][(df * 16 + n) * 32], g);
      accO[df] = wmma_bf16(ap, bv, accO[df]);
    }
    __syncthreads();  // all reads of buf t done before TDM reuses it
  }

  // finalize: l = rowsum, divide, store context [B*S, D] bf16
  float linv[8];
  for (int r = 0; r < 8; ++r) {
    float v = lrun[r];
    for (int off = 1; off < 16; off <<= 1) v += __shfl_xor(v, off, 32);
    linv[r] = 1.0f / v;
  }
  for (int df = 0; df < 4; ++df)
    for (int r = 0; r < 8; ++r) {
      long row = (long)b * Sc + q0 + 8 * g + r;
      int col = h * DKc + df * 16 + n;
      ctx[row * Dc + col] = (bf16)(accO[df][r] * linv[r]);
    }
}

// ---------------------------------------------------------------------------
// Kernel 3: output projection (Wt_o pre-transposed, TDM slabs) + residual +
// LayerNorm fused.  Workgroup spans full 1024-wide rows -> LN reduce stays
// in-workgroup (shfl + ds_add_f32 atomics).
// ---------------------------------------------------------------------------
__global__ __launch_bounds__(256)
void out_proj_ln_kernel(const bf16* __restrict__ Actx, const bf16* __restrict__ Wt,
                        const float* __restrict__ resid,
                        const float* __restrict__ gamma,
                        const float* __restrict__ beta,
                        float* __restrict__ out) {
  __shared__ __align__(16) bf16 wt[2][Dc * 32];
  __shared__ float red_s[32], red_q[32];
  const int tid = threadIdx.x;
  const int wid = tid >> 5;
  const int lane = tid & 31;
  const int g = lane >> 4, n = lane & 15;
  const int rowbase = blockIdx.x * 32;
  const int colw = wid * 128;

  f32x8 acc[2][8];
  for (int i = 0; i < 2; ++i)
    for (int j = 0; j < 8; ++j)
      for (int r = 0; r < 8; ++r) acc[i][j][r] = 0.f;

  constexpr int T = Dc / 32;
  if (wid == 0)
    tdm_load_2d(lds_addr_of(&wt[0][0]), Wt, 1, 32, Dc, Dc);

  for (int t = 0; t < T; ++t) {
    const int kk = t * 32;
    if (wid == 0) {
      if (t + 1 < T) {
        tdm_load_2d(lds_addr_of(&wt[(t + 1) & 1][0]), Wt + kk + 32, 1, 32, Dc, Dc);
        __builtin_amdgcn_s_wait_tensorcnt(1);
      } else {
        __builtin_amdgcn_s_wait_tensorcnt(0);
      }
    }
    __syncthreads();

    const bf16* wbuf = &wt[t & 1][0];
    bf16x16 af[2];
    for (int fm = 0; fm < 2; ++fm) {
      const bf16* p = Actx + (long)(rowbase + fm * 16 + n) * Dc + kk;
      af[fm] = fragA(p, g);
    }
    for (int fn = 0; fn < 8; ++fn) {
      bf16x16 bfg = fragB(wbuf + (colw + fn * 16 + n) * 32, g);
      acc[0][fn] = wmma_bf16(af[0], bfg, acc[0][fn]);
      acc[1][fn] = wmma_bf16(af[1], bfg, acc[1][fn]);
    }
    __syncthreads();
  }

  // residual add
  for (int fm = 0; fm < 2; ++fm)
    for (int fn = 0; fn < 8; ++fn)
      for (int r = 0; r < 8; ++r) {
        int row = rowbase + fm * 16 + 8 * g + r;
        int col = colw + fn * 16 + n;
        acc[fm][fn][r] += resid[(long)row * Dc + col];
      }

  if (tid < 32) { red_s[tid] = 0.f; red_q[tid] = 0.f; }
  __syncthreads();
  for (int fm = 0; fm < 2; ++fm)
    for (int r = 0; r < 8; ++r) {
      float s = 0.f, q = 0.f;
      for (int fn = 0; fn < 8; ++fn) {
        float v = acc[fm][fn][r];
        s += v; q += v * v;
      }
      for (int off = 1; off < 16; off <<= 1) {
        s += __shfl_xor(s, off, 32);
        q += __shfl_xor(q, off, 32);
      }
      if (n == 0) {
        int lr = fm * 16 + 8 * g + r;
        atomicAdd(&red_s[lr], s);   // ds_add_f32
        atomicAdd(&red_q[lr], q);
      }
    }
  __syncthreads();
  for (int fm = 0; fm < 2; ++fm)
    for (int r = 0; r < 8; ++r) {
      int lr = fm * 16 + 8 * g + r;
      int row = rowbase + lr;
      float mu = red_s[lr] * (1.0f / 1024.0f);
      float var = red_q[lr] * (1.0f / 1024.0f) - mu * mu;
      float rstd = rsqrtf(var + 1e-6f);
      for (int fn = 0; fn < 8; ++fn) {
        int col = colw + fn * 16 + n;
        out[(long)row * Dc + col] =
            (acc[fm][fn][r] - mu) * rstd * gamma[col] + beta[col];
      }
    }
}

// ---------------------------------------------------------------------------
// Launch
// ---------------------------------------------------------------------------
extern "C" void kernel_launch(void* const* d_in, const int* in_sizes, int n_in,
                              void* d_out, int out_size, void* d_ws, size_t ws_size,
                              hipStream_t stream) {
  const float* q     = (const float*)d_in[0];
  const float* k     = (const float*)d_in[1];
  const float* v     = (const float*)d_in[2];
  const float* mask  = (const float*)d_in[3];
  const float* Wq    = (const float*)d_in[4];
  const float* Wk    = (const float*)d_in[5];
  const float* Wv    = (const float*)d_in[6];
  const float* Wo    = (const float*)d_in[7];
  const float* gamma = (const float*)d_in[8];
  const float* beta  = (const float*)d_in[9];
  float* out = (float*)d_out;

  const long nAct = (long)Bc * Sc * Dc;  // 8388608
  const long nW   = (long)Dc * Dc;       // 1048576

  bf16* base = (bf16*)d_ws;
  bf16* qb   = base;               // bf16 copies of activations
  bf16* kb   = qb  + nAct;
  bf16* vb   = kb  + nAct;
  bf16* wtq  = vb  + nAct;         // Wt = W^T, bf16 [N][K]
  bf16* wtk  = wtq + nW;
  bf16* wtv  = wtk + nW;
  bf16* wto  = wtv + nW;
  bf16* Qh   = wto + nW;           // [B,H,S,dk], pre-scaled by 1/sqrt(dk)
  bf16* Kh   = Qh  + nAct;         // [B,H,S,dk]
  bf16* Vt   = Kh  + nAct;         // [B,H,dk,S]
  bf16* Ctx  = Vt  + nAct;         // [B*S, D]

  cvt_f32_to_bf16<<<2048, 256, 0, stream>>>(q, qb, nAct);
  cvt_f32_to_bf16<<<2048, 256, 0, stream>>>(k, kb, nAct);
  cvt_f32_to_bf16<<<2048, 256, 0, stream>>>(v, vb, nAct);
  dim3 tg(Dc / 32, Dc / 32);
  transpose_w_kernel<<<tg, 256, 0, stream>>>(Wq, wtq);
  transpose_w_kernel<<<tg, 256, 0, stream>>>(Wk, wtk);
  transpose_w_kernel<<<tg, 256, 0, stream>>>(Wv, wtv);
  transpose_w_kernel<<<tg, 256, 0, stream>>>(Wo, wto);

  dim3 pgrid((Bc * Sc) / 128, Dc / 256);  // 64 x 4
  proj_gemm_kernel<<<pgrid, 256, 0, stream>>>(qb, wtq, Qh, 0.125f, 0);
  proj_gemm_kernel<<<pgrid, 256, 0, stream>>>(kb, wtk, Kh, 1.0f, 0);
  proj_gemm_kernel<<<pgrid, 256, 0, stream>>>(vb, wtv, Vt, 1.0f, 1);

  flash_attn_kernel<<<dim3(Sc / 64, Bc * Hc), 128, 0, stream>>>(Qh, Kh, Vt, mask, Ctx);

  out_proj_ln_kernel<<<(Bc * Sc) / 32, 256, 0, stream>>>(Ctx, wto, q, gamma, beta, out);
}
